// KernelAggregation_59940563583649
// MI455X (gfx1250) — matrix operand order repaired and checked by reading, not compiled, not run on hardware
//
#include <hip/hip_runtime.h>
#include <hip/hip_bf16.h>

typedef __attribute__((ext_vector_type(16))) _Float16 v16h;
typedef __attribute__((ext_vector_type(8)))  _Float16 v8h;
typedef __attribute__((ext_vector_type(8)))  float    v8f;
typedef unsigned int u32x4 __attribute__((ext_vector_type(4)));
typedef int          i32x8 __attribute__((ext_vector_type(8)));
typedef int          i32x4 __attribute__((ext_vector_type(4)));

#define B_    16
#define C_    256
#define H_    56
#define W_    56
#define HW_   3136
#define NK    8
#define KTAPS 9
#define KTOT  (C_ * KTAPS)   // 2304, K order: k = tap*256 + ci
#define NCHUNK 72            // KTOT / 32

#define MT  256              // Cout tile = full Cout
#define NT  64               // pixel tile
#define KC  32               // K chunk
#define LDA 40               // padded f16 row stride (80 B, 16B-aligned rows)
#define LDB 40

// ---------------------------------------------------------------------------
// Kernel 1: per-sample weight mixing into f16, tap-major K layout.
// wmix[b][co][tap][ci] = sum_n att[b,n] * W[n, co, ci, tap]
// ---------------------------------------------------------------------------
__global__ __launch_bounds__(256)
void mix_weights(const float* __restrict__ att,
                 const float* __restrict__ wgt,
                 _Float16* __restrict__ wmix) {
  __shared__ float a_sh[B_ * NK];
  const int t = threadIdx.x;
  if (t < B_ * NK) a_sh[t] = att[t];
  __syncthreads();

  const int e  = blockIdx.x * 256 + t;   // 0 .. 256*2304-1 (co * ktot)
  const int co = e / KTOT;
  const int kp = e % KTOT;               // tap*256 + ci
  const int tap = kp / C_;
  const int ci  = kp % C_;

  float wv[NK];
#pragma unroll
  for (int n = 0; n < NK; ++n)
    wv[n] = wgt[(((size_t)n * C_ + co) * C_ + ci) * KTAPS + tap];

#pragma unroll
  for (int b = 0; b < B_; ++b) {
    float s = 0.f;
#pragma unroll
    for (int n = 0; n < NK; ++n) s += a_sh[b * NK + n] * wv[n];
    wmix[(size_t)b * C_ * KTOT + e] = (_Float16)s;   // coalesced
  }
}

// Build a v16h fragment from two 16-byte LDS reads.
__device__ __forceinline__ v16h ld_frag(const _Float16* p0, const _Float16* p1) {
  v8h lo = *(const v8h*)p0;
  v8h hi = *(const v8h*)p1;
  v16h r;
#pragma unroll
  for (int i = 0; i < 8; ++i) { r[i] = lo[i]; r[i + 8] = hi[i]; }
  return r;
}

// ---------------------------------------------------------------------------
// TDM: load one A tile (256 rows x 32 f16, row stride KTOT f16) from wmix
// into LDS at laddr, writing padded 80-byte rows (64B data + 16B pad).
// D# per CDNA5 ISA ch.8 (group0: count/lds/global/type; group1: dims/pad).
// ---------------------------------------------------------------------------
__device__ __forceinline__ void tdm_load_a(unsigned long long gaddr,
                                           unsigned int laddr) {
  u32x4 g0;
  g0[0] = 1u;                                       // count=1, user descriptor
  g0[1] = laddr;                                    // lds_addr (bytes)
  g0[2] = (unsigned int)gaddr;                      // global_addr[31:0]
  g0[3] = (unsigned int)(gaddr >> 32) | (2u << 30); // addr[56:32] | type=2
  i32x8 g1;
  g1[0] = (1 << 16)      // data_size = 2 bytes
        | (1 << 20)      // pad_enable
        | (3 << 22)      // pad_interval: 16 DWORDs = 64 B
        | (3 << 25);     // pad_amount:   4 DWORDs = 16 B
  g1[1] = (int)((KTOT & 0xFFFF) << 16);             // tensor_dim0[15:0]
  g1[2] = (int)((KTOT >> 16) | (MT << 16));         // dim0[31:16] | tensor_dim1
  g1[3] = (int)(KC << 16);                          // tile_dim0 = 32
  g1[4] = MT;                                       // tile_dim1 = 256 rows
  g1[5] = KTOT;                                     // tensor_dim0_stride
  g1[6] = 0;
  g1[7] = 0;
  i32x4 z4 = {};
#if __clang_major__ >= 23
  i32x8 z8 = {};
  __builtin_amdgcn_tensor_load_to_lds(g0, g1, z4, z4, z8, 0);
#else
  __builtin_amdgcn_tensor_load_to_lds(g0, g1, z4, z4, 0);
#endif
}

// Branchless B prefetch: 8 consecutive ci planes for this thread's pixel.
__device__ __forceinline__ void load_bregs(const float* __restrict__ xb,
                                           int ph, int pw, int kc, int kq,
                                           float* r, float* msk) {
  const int tap = kc >> 3;
  const int cc  = kc & 7;
  const int dh = tap / 3 - 1;
  const int dw = tap % 3 - 1;
  const int y  = ph + dh;
  const int xx = pw + dw;
  *msk = ((y >= 0) & (y < H_) & (xx >= 0) & (xx < W_)) ? 1.f : 0.f;
  const int yc = min(max(y, 0), H_ - 1);
  const int xc = min(max(xx, 0), W_ - 1);
  const float* base = xb + (size_t)(cc * KC + kq * 8) * HW_ + yc * W_ + xc;
#pragma unroll
  for (int i = 0; i < 8; ++i) r[i] = base[(size_t)i * HW_];
}

// ---------------------------------------------------------------------------
// Kernel 2: implicit-GEMM dynamic conv, double-buffered, TDM-fed A tiles.
// Block: 256 thr (8 wave32), tile 256(co) x 64(px), K chunk 32.
// Each wave owns 32 co rows x 64 px: 2 A-frags x 4 B-frags = 8 WMMA/chunk.
// ---------------------------------------------------------------------------
__global__ __launch_bounds__(256)
void dynconv_gemm(const float* __restrict__ x,
                  const float* __restrict__ att,
                  const float* __restrict__ bias,
                  const _Float16* __restrict__ wmix,
                  float* __restrict__ out) {
  __shared__ _Float16 a_s[2][MT * LDA];
  __shared__ _Float16 b_s[2][NT * LDB];
  __shared__ float    bmix_s[MT];

  const int t  = threadIdx.x;
  const int b  = blockIdx.z;
  const int n0 = blockIdx.x * NT;

  {
    float s = 0.f;
#pragma unroll
    for (int n = 0; n < NK; ++n) s += att[b * NK + n] * bias[n * C_ + t];
    bmix_s[t] = s;
  }

  // B fill: thread owns pixel nn, 8 contiguous ci slots starting at kq*8
  const int nn = t & 63;
  const int kq = t >> 6;                 // 0..3
  const int p  = n0 + nn;
  const int ph = p / W_;
  const int pw = p % W_;
  const float* xb = x + (size_t)b * C_ * HW_;
  const unsigned long long wb_addr =
      (unsigned long long)(uintptr_t)(wmix + (size_t)b * C_ * KTOT);

  const int lane = t & 31;
  const int wid  = t >> 5;
  const int hi_half = (lane & 16) ? 1 : 0;
  const int l15  = lane & 15;

  v8f acc[2][4] = {};

  // prologue: TDM chunk 0 -> a_s[0]; B regs chunk 0
  if (wid == 0)
    tdm_load_a(wb_addr, (unsigned int)(uintptr_t)(&a_s[0][0]));
  float r[8], msk;
  load_bregs(xb, ph, pw, 0, kq, r, &msk);

#pragma unroll 1
  for (int kc = 0; kc < NCHUNK; ++kc) {
    const int pb = kc & 1;

    // pack + single ds_store_b128 of this thread's 8 halves
    v8h hv;
#pragma unroll
    for (int i = 0; i < 8; ++i) hv[i] = (_Float16)(r[i] * msk);
    *(v8h*)(&b_s[pb][nn * LDB + kq * 8]) = hv;

    __builtin_amdgcn_s_wait_tensorcnt(0);  // scalar; free on non-issuing waves
    __syncthreads();                       // dscnt waited by compiler

    if (kc + 1 < NCHUNK) {
      if (wid == 0)
        tdm_load_a(wb_addr + (unsigned long long)(kc + 1) * (KC * 2),
                   (unsigned int)(uintptr_t)(&a_s[pb ^ 1][0]));
      load_bregs(xb, ph, pw, kc + 1, kq, r, &msk);   // in flight under WMMAs
    }

    // Hoist all fragment loads, then burst 8 independent WMMAs.
    const int arow  = wid * 32 + l15;
    const int abase = hi_half * 8;
    const v16h afr0 = ld_frag(&a_s[pb][arow * LDA + abase],
                              &a_s[pb][arow * LDA + abase + 16]);
    const v16h afr1 = ld_frag(&a_s[pb][(arow + 16) * LDA + abase],
                              &a_s[pb][(arow + 16) * LDA + abase + 16]);
    v16h bfr[4];
#pragma unroll
    for (int tt = 0; tt < 4; ++tt) {
      const int brow = tt * 16 + l15;
      const int kb   = hi_half * 8;        // K halves kb..kb+7, kb+16..kb+23
      bfr[tt] = ld_frag(&b_s[pb][brow * LDB + kb],
                        &b_s[pb][brow * LDB + kb + 16]);
    }
#pragma unroll
    for (int tt = 0; tt < 4; ++tt) {
      acc[0][tt] = __builtin_amdgcn_wmma_f32_16x16x32_f16(
          false, afr0, false, bfr[tt], (short)0, acc[0][tt], false, false);
      acc[1][tt] = __builtin_amdgcn_wmma_f32_16x16x32_f16(
          false, afr1, false, bfr[tt], (short)0, acc[1][tt], false, false);
    }
  }

  // epilogue: C layout VGPR v -> M = v + 8*hi_half, N = lane%16
  float* ob = out + (size_t)b * C_ * HW_ + n0;
#pragma unroll
  for (int a = 0; a < 2; ++a) {
#pragma unroll
    for (int tt = 0; tt < 4; ++tt) {
#pragma unroll
      for (int v = 0; v < 8; ++v) {
        const int m = wid * 32 + a * 16 + v + 8 * hi_half;
        ob[(size_t)m * HW_ + tt * 16 + l15] = acc[a][tt][v] + bmix_s[m];
      }
    }
  }
}

extern "C" void kernel_launch(void* const* d_in, const int* in_sizes, int n_in,
                              void* d_out, int out_size, void* d_ws, size_t ws_size,
                              hipStream_t stream) {
  const float* x    = (const float*)d_in[0];   // (16,256,56,56)
  const float* att  = (const float*)d_in[1];   // (16,8)
  const float* wgt  = (const float*)d_in[2];   // (8,256,256,3,3)
  const float* bias = (const float*)d_in[3];   // (8,256)
  float* out = (float*)d_out;                  // (16,256,56,56)
  _Float16* wmix = (_Float16*)d_ws;            // 16*256*2304 f16 = 18.9 MB

  mix_weights<<<dim3(KTOT * C_ / 256), dim3(256), 0, stream>>>(att, wgt, wmix);
  dynconv_gemm<<<dim3(HW_ / NT, 1, B_), dim3(256), 0, stream>>>(
      x, att, bias, wmix, out);
}
